// NonLocalAttention_29910152250053
// MI455X (gfx1250) — matrix operand order repaired and checked by reading, not compiled, not run on hardware
//
#include <hip/hip_runtime.h>

#define HWD 6400
#define CD  128
#define CED 64
#define NJC (HWD / 32)

typedef _Float16 v8h  __attribute__((ext_vector_type(8)));
typedef _Float16 v16h __attribute__((ext_vector_type(16)));
typedef _Float16 h2   __attribute__((ext_vector_type(2)));
typedef float    v8f  __attribute__((ext_vector_type(8)));
typedef unsigned v8u  __attribute__((ext_vector_type(8)));

// ---- CDNA5 async global->LDS path (guarded; falls back to reg pipeline) ----
#if defined(__has_builtin)
# if __has_builtin(__builtin_amdgcn_global_load_async_to_lds_b128)
#  define HAS_ASYNC_LDS 1
# endif
#endif
#ifndef HAS_ASYNC_LDS
# define HAS_ASYNC_LDS 0
#endif

#if HAS_ASYNC_LDS
// builtin signature (from hipcc diagnostic): (int4 AS1*, int4 AS3*, Ii, Ii)
typedef int i4 __attribute__((__vector_size__(16)));
typedef __attribute__((address_space(1))) i4 as1_i4;
typedef __attribute__((address_space(3))) i4 as3_i4;

__device__ __forceinline__ void async_b128(const _Float16* g, _Float16* l) {
    __builtin_amdgcn_global_load_async_to_lds_b128(
        (as1_i4*)g, (as3_i4*)l, 0, 0);
}
__device__ __forceinline__ void wait_async0() {
#if __has_builtin(__builtin_amdgcn_s_wait_asynccnt)
    __builtin_amdgcn_s_wait_asynccnt(0);
#else
    asm volatile("s_wait_asynccnt 0x0" ::: "memory");
#endif
}
#endif

// lane XOR 16 exchange (SWAPX16): group-of-32 swizzle, xor=0x10, and=0x1f
__device__ __forceinline__ float swz16f(float x) {
    return __builtin_bit_cast(float,
        __builtin_amdgcn_ds_swizzle(__builtin_bit_cast(int, x), 0x401F));
}
__device__ __forceinline__ unsigned swz16u(unsigned x) {
    return (unsigned)__builtin_amdgcn_ds_swizzle((int)x, 0x401F);
}
__device__ __forceinline__ unsigned pk16(float a, float b) {
    h2 t; t.x = (_Float16)a; t.y = (_Float16)b;
    return __builtin_bit_cast(unsigned, t);
}
__device__ __forceinline__ v16h cat8(v8h lo, v8h hi) {
    return __builtin_shufflevector(lo, hi, 0,1,2,3,4,5,6,7,8,9,10,11,12,13,14,15);
}
__device__ __forceinline__ v8f wmma16(v16h a, v16h b, v8f c) {
    return __builtin_amdgcn_wmma_f32_16x16x32_f16(false, a, false, b,
                                                  (short)0, c, false, false);
}
// A-operand from fp32 weight row: halves 0..7 = base[0..7], 8..15 = base[16..23]
// (base already includes +cb*32 + 8*hi, matching the ISA 16-bit A layout)
__device__ __forceinline__ v16h mkA(const float* base) {
    float4 a = ((const float4*)base)[0];
    float4 b = ((const float4*)base)[1];
    float4 c = ((const float4*)(base + 16))[0];
    float4 d = ((const float4*)(base + 16))[1];
    v8u u;
    u[0] = pk16(a.x, a.y); u[1] = pk16(a.z, a.w);
    u[2] = pk16(b.x, b.y); u[3] = pk16(b.z, b.w);
    u[4] = pk16(c.x, c.y); u[5] = pk16(c.z, c.w);
    u[6] = pk16(d.x, d.y); u[7] = pk16(d.z, d.w);
    return __builtin_bit_cast(v16h, u);
}

// ---------------------------------------------------------------------------
// Kernel 1: 1x1 conv + PReLU via WMMA.  Block = 256 thr, 32 hw positions,
// all 256 out-channels (q0..63 | k0..63 | v0..127).  Wave w owns oc 32w..32w+31
// (2 oc-tiles) x 2 hw-tiles; K=128 in 4 chunks -> 16 wmma/wave.
//   q16[n][hw][64], k16[n][hw][64], v16[n][c][hw]   (f16)
// ---------------------------------------------------------------------------
__global__ __launch_bounds__(256) void proj_kernel(
    const float* __restrict__ x,
    const float* __restrict__ Wq, const float* __restrict__ bq, const float* __restrict__ aq,
    const float* __restrict__ Wk, const float* __restrict__ bk, const float* __restrict__ ak,
    const float* __restrict__ Wv, const float* __restrict__ bv, const float* __restrict__ av,
    _Float16* __restrict__ q16, _Float16* __restrict__ k16, _Float16* __restrict__ v16)
{
    __shared__ _Float16 xs[32][152];   // [hw][c] transposed, pad 128->152 (16B rows)

    const int tid = threadIdx.x;
    const int lane = tid & 31, w = tid >> 5;
    const int m16 = lane & 15, hi = lane >> 4;
    const int n   = blockIdx.x / 200;
    const int hw0 = (blockIdx.x % 200) * 32;

    // stage x tile [128c x 32hw] -> xs[hw][c] with f32->f16 convert
#pragma unroll
    for (int it = 0; it < 16; ++it) {
        int idx = tid + it * 256;
        int hwl = idx & 31, c = idx >> 5;
        xs[hwl][c] = (_Float16)x[((size_t)(n * CD) + c) * HWD + hw0 + hwl];
    }

    // per-wave family selection (wave-uniform: w<2 q, w<4 k, else v)
    const int oc0 = w * 32;
    const int fam_base = (w < 2) ? 0 : (w < 4) ? 64 : 128;
    const float* Wf = (w < 2) ? Wq : (w < 4) ? Wk : Wv;
    const float* bf = (w < 2) ? bq : (w < 4) ? bk : bv;
    const float slope = (w < 2) ? aq[0] : (w < 4) ? ak[0] : av[0];
    const int row0 = oc0 - fam_base;

    // W A-operands: A[t][cb], rows row0+16t+m16, K-chunk cb (f32->f16 in regs)
    v16h A[2][4];
#pragma unroll
    for (int t = 0; t < 2; ++t) {
        const float* wr = Wf + (size_t)(row0 + 16 * t + m16) * CD + 8 * hi;
#pragma unroll
        for (int cb = 0; cb < 4; ++cb) A[t][cb] = mkA(wr + cb * 32);
    }

    __syncthreads();

    const v8f zerov = {0.f,0.f,0.f,0.f,0.f,0.f,0.f,0.f};
    v8f D[2][2] = {{zerov, zerov}, {zerov, zerov}};
#pragma unroll
    for (int cb = 0; cb < 4; ++cb) {
        // B = x^T chunk: lane m16 = hw, elems 0..7 c 8*hi.., 8..15 c 16+8*hi..
        const _Float16* b0 = &xs[m16][cb * 32 + 8 * hi];
        const _Float16* b1 = &xs[16 + m16][cb * 32 + 8 * hi];
        v16h B0 = cat8(((const v8h*)b0)[0], ((const v8h*)(b0 + 16))[0]);
        v16h B1 = cat8(((const v8h*)b1)[0], ((const v8h*)(b1 + 16))[0]);
        D[0][0] = wmma16(A[0][cb], B0, D[0][0]);
        D[0][1] = wmma16(A[0][cb], B1, D[0][1]);
        D[1][0] = wmma16(A[1][cb], B0, D[1][0]);
        D[1][1] = wmma16(A[1][cb], B1, D[1][1]);
    }

    _Float16* dst = (w < 2) ? q16 : (w < 4) ? k16 : v16;
#pragma unroll
    for (int t = 0; t < 2; ++t)
#pragma unroll
    for (int h = 0; h < 2; ++h)
#pragma unroll
    for (int r = 0; r < 8; ++r) {
        int loc = row0 + 16 * t + r + 8 * hi;       // channel within family
        int hw  = hw0 + 16 * h + m16;
        float y = D[t][h][r] + bf[loc];
        y = (y >= 0.f) ? y : slope * y;
        if (w < 4)   // q/k: [n][hw][64]
            dst[((size_t)(n * HWD) + hw) * CED + loc] = (_Float16)y;
        else         // v: [n][c][hw]
            dst[((size_t)(n * CD) + loc) * HWD + hw] = (_Float16)y;
    }
}

// ---------------------------------------------------------------------------
// Kernel 2: fused flash attention, double-buffered K/V staging (async->LDS).
// Block = 8 waves x 16 i-rows. S^T = K x Q^T; O += V x P^T.
// ---------------------------------------------------------------------------
__global__ __launch_bounds__(256) void attn_kernel(
    const _Float16* __restrict__ q16, const _Float16* __restrict__ k16,
    const _Float16* __restrict__ v16, float* __restrict__ out)
{
    __shared__ _Float16 ks[2][32][72];    // j_local x ce (pad 64->72)
    __shared__ _Float16 vs[2][128][40];   // c x j_local  (pad 32->40)

    const int tid  = threadIdx.x;
    const int lane = tid & 31, w = tid >> 5;
    const int m16  = lane & 15, hi = lane >> 4;
    const int n    = blockIdx.x / 50;
    const int i0   = (blockIdx.x % 50) * 128 + w * 16;

    const v8f zerov = {0.f,0.f,0.f,0.f,0.f,0.f,0.f,0.f};

    const _Float16* kbase = k16 + (size_t)n * HWD * CED;
    const _Float16* vbase = v16 + (size_t)n * CD * HWD;

    // per-thread staging slots
    const int krow = tid >> 3, kseg = tid & 7;     // K: 32 rows x 8 segs of 8 halves
    const int vc   = tid >> 1, vhh = tid & 1;      // V: 128 rows x 2 segs of 16 halves
    const _Float16* ksrc0 = kbase + (size_t)krow * CED + kseg * 8;
    const _Float16* vsrc0 = vbase + (size_t)vc * HWD + vhh * 16;

    // Q^T B-operands (fixed per wave)
    const _Float16* qb = q16 + ((size_t)(n * HWD) + i0 + m16) * CED + 8 * hi;
    v16h BQ0 = cat8(((const v8h*)qb)[0], ((const v8h*)(qb + 16))[0]);
    v16h BQ1 = cat8(((const v8h*)(qb + 32))[0], ((const v8h*)(qb + 48))[0]);

    v8f oacc[8];
#pragma unroll
    for (int t = 0; t < 8; ++t) oacc[t] = zerov;
    float m_run = -3.4e38f, l_run = 0.f;

#if HAS_ASYNC_LDS
    async_b128(ksrc0,     &ks[0][krow][kseg * 8]);
    async_b128(vsrc0,     &vs[0][vc][vhh * 16]);
    async_b128(vsrc0 + 8, &vs[0][vc][vhh * 16 + 8]);
#else
    v8h kreg = *(const v8h*)ksrc0;
    v8h vreg0 = ((const v8h*)vsrc0)[0];
    v8h vreg1 = ((const v8h*)vsrc0)[1];
#endif

    for (int jc = 0; jc < NJC; ++jc) {
        const int buf = jc & 1;
#if HAS_ASYNC_LDS
        wait_async0();
        __syncthreads();
        if (jc + 1 < NJC) {
            const _Float16* kn = ksrc0 + (size_t)(jc + 1) * 32 * CED;
            const _Float16* vn = vsrc0 + (jc + 1) * 32;
            async_b128(kn,     &ks[1 - buf][krow][kseg * 8]);
            async_b128(vn,     &vs[1 - buf][vc][vhh * 16]);
            async_b128(vn + 8, &vs[1 - buf][vc][vhh * 16 + 8]);
        }
#else
        *(v8h*)&ks[buf][krow][kseg * 8]   = kreg;
        *(v8h*)&vs[buf][vc][vhh * 16]     = vreg0;
        *(v8h*)&vs[buf][vc][vhh * 16 + 8] = vreg1;
        if (jc + 1 < NJC) {
            const _Float16* kn = ksrc0 + (size_t)(jc + 1) * 32 * CED;
            const _Float16* vn = vsrc0 + (jc + 1) * 32;
            kreg  = *(const v8h*)kn;
            vreg0 = ((const v8h*)vn)[0];
            vreg1 = ((const v8h*)vn)[1];
        }
        __syncthreads();
#endif
        // S^T tiles (A = K rows in ISA A-layout, straight from LDS)
        v16h A0, A1;
        A0 = cat8(*(const v8h*)&ks[buf][m16][8*hi],      *(const v8h*)&ks[buf][m16][16 + 8*hi]);
        A1 = cat8(*(const v8h*)&ks[buf][m16][32 + 8*hi], *(const v8h*)&ks[buf][m16][48 + 8*hi]);
        v8f sa = wmma16(A0, BQ0, zerov);
        sa = wmma16(A1, BQ1, sa);
        A0 = cat8(*(const v8h*)&ks[buf][16+m16][8*hi],      *(const v8h*)&ks[buf][16+m16][16 + 8*hi]);
        A1 = cat8(*(const v8h*)&ks[buf][16+m16][32 + 8*hi], *(const v8h*)&ks[buf][16+m16][48 + 8*hi]);
        v8f sb = wmma16(A0, BQ0, zerov);
        sb = wmma16(A1, BQ1, sb);

        // online softmax (row i = lane&15; halves paired via xor16 swizzle)
        float mx = sa[0];
#pragma unroll
        for (int r = 1; r < 8; ++r) mx = fmaxf(mx, sa[r]);
#pragma unroll
        for (int r = 0; r < 8; ++r) mx = fmaxf(mx, sb[r]);
        mx = fmaxf(mx, swz16f(mx));
        float m_new = fmaxf(m_run, mx);
        float f = __expf(m_run - m_new);
        float pa[8], pb[8], sum = 0.f;
#pragma unroll
        for (int r = 0; r < 8; ++r) {
            pa[r] = __expf(sa[r] - m_new); sum += pa[r];
            pb[r] = __expf(sb[r] - m_new); sum += pb[r];
        }
        sum += swz16f(sum);
        l_run = l_run * f + sum;
        m_run = m_new;
#pragma unroll
        for (int t = 0; t < 8; ++t) oacc[t] = oacc[t] * f;

        // pack P^T as f16 B operand (low lanes K=0..15, high lanes K=16..31)
        unsigned pau[4], pbu[4];
#pragma unroll
        for (int q = 0; q < 4; ++q) {
            pau[q] = pk16(pa[2*q], pa[2*q+1]);
            pbu[q] = pk16(pb[2*q], pb[2*q+1]);
        }
        v8u bpu;
#pragma unroll
        for (int q = 0; q < 4; ++q) {
            unsigned oa = swz16u(pau[q]);
            unsigned ob = swz16u(pbu[q]);
            bpu[q]     = hi ? ob     : pau[q];
            bpu[4 + q] = hi ? pbu[q] : oa;
        }
        v16h BP = __builtin_bit_cast(v16h, bpu);

        // O[c-tile] += V-rows x P^T
#pragma unroll
        for (int ct = 0; ct < 8; ++ct) {
            int c = ct * 16 + m16;
            v16h AV = cat8(*(const v8h*)&vs[buf][c][16*hi],
                           *(const v8h*)&vs[buf][c][16*hi + 8]);
            oacc[ct] = wmma16(AV, BP, oacc[ct]);
        }
    }

    const float inv = 1.f / l_run;
#pragma unroll
    for (int ct = 0; ct < 8; ++ct)
#pragma unroll
        for (int r = 0; r < 8; ++r)
            out[((size_t)(n * CD + ct * 16 + r + 8 * hi)) * HWD + i0 + m16] =
                oacc[ct][r] * inv;
}

// ---------------------------------------------------------------------------
extern "C" void kernel_launch(void* const* d_in, const int* in_sizes, int n_in,
                              void* d_out, int out_size, void* d_ws, size_t ws_size,
                              hipStream_t stream) {
    (void)in_sizes; (void)n_in; (void)out_size; (void)ws_size;
    const float* x  = (const float*)d_in[0];
    const float* Wq = (const float*)d_in[1];
    const float* bq = (const float*)d_in[2];
    const float* aq = (const float*)d_in[3];
    const float* Wk = (const float*)d_in[4];
    const float* bk = (const float*)d_in[5];
    const float* ak = (const float*)d_in[6];
    const float* Wv = (const float*)d_in[7];
    const float* bv = (const float*)d_in[8];
    const float* av = (const float*)d_in[9];

    _Float16* q16 = (_Float16*)d_ws;                 // 4*6400*64 halves
    _Float16* k16 = q16 + (size_t)4 * HWD * CED;
    _Float16* v16 = k16 + (size_t)4 * HWD * CED;     // 4*128*6400 halves
    float* out = (float*)d_out;

    proj_kernel<<<800, 256, 0, stream>>>(x, Wq, bq, aq, Wk, bk, ak, Wv, bv, av,
                                         q16, k16, v16);
    attn_kernel<<<200, 256, 0, stream>>>(q16, k16, v16, out);
}
